// MambaModule_58239756534202
// MI455X (gfx1250) — matrix operand order repaired and checked by reading, not compiled, not run on hardware
//
#include <hip/hip_runtime.h>
#include <hip/hip_bf16.h>
#include <math.h>

typedef __attribute__((ext_vector_type(16))) _Float16 v16h;
typedef __attribute__((ext_vector_type(8)))  float    v8f;
typedef __attribute__((ext_vector_type(4)))  float    v4f;

#define LSEQ   4096
#define DIMC   256
#define DINNER 512
#define DSTATE 16
#define NBATCH 8
#define NCHUNK 64
#define CLEN   64          // LSEQ / NCHUNK
#define XSTR   64          // padded xdbc row stride (dt:0-15, B:16-31, C:32-47)

// ---------------------------------------------------------------------------
// Tiled transpose: x_feat (B, C, L) -> seq0 (B, L, C)
// ---------------------------------------------------------------------------
__global__ __launch_bounds__(256)
void transpose_bcl_to_blc(const float* __restrict__ x, float* __restrict__ s)
{
    __shared__ float tile[32][33];
    int b  = blockIdx.z;
    int c0 = blockIdx.y * 32;
    int l0 = blockIdx.x * 32;
    for (int i = threadIdx.y; i < 32; i += 8)
        tile[i][threadIdx.x] =
            x[((size_t)b * DIMC + c0 + i) * LSEQ + l0 + threadIdx.x];
    __syncthreads();
    for (int i = threadIdx.y; i < 32; i += 8)
        s[((size_t)b * LSEQ + l0 + i) * DIMC + c0 + threadIdx.x] =
            tile[threadIdx.x][i];
}

// ---------------------------------------------------------------------------
// Weight padding (once per launch; removes all per-lane guards in the GEMM)
// ---------------------------------------------------------------------------
__global__ __launch_bounds__(256)
void pad_xproj_w(const float* __restrict__ src, float* __restrict__ dst)
{   // 48x512 -> 64x512, zero rows 48..63
    int idx = blockIdx.x * 256 + threadIdx.x;          // 64*512
    int r = idx >> 9, c = idx & 511;
    dst[idx] = (r < 48) ? src[r * 512 + c] : 0.0f;
}

__global__ __launch_bounds__(256)
void pad_dt_w(const float* __restrict__ src, float* __restrict__ dst)
{   // 512x16 -> 512x32, zero cols 16..31
    int idx = blockIdx.x * 256 + threadIdx.x;          // 512*32
    int r = idx >> 5, c = idx & 31;
    dst[idx] = (c < 16) ? src[r * 16 + c] : 0.0f;
}

// ---------------------------------------------------------------------------
// WMMA GEMM: C[m,n] = sum_k A[m,k] * W[n,k]  (+bias, epilogue modes)
//   MODE 0: C[m*ldc+n] = v (+bias)
//   MODE 1: softplus(v + bias)                                   (dt_proj)
//   MODE 2: bias, store out[b, n, l], b=m/LSEQ, l=m%LSEQ          (final)
// Requirements (guaranteed by caller): K % 32 == 0, M % 32 == 0,
// N % 64 == 0 (padded), all row strides % 4 == 0 (16B-aligned v4f loads).
// Block = 128 threads (4 waves). Wave tile: 32x16 (2 WMMAs share B frag).
// Fragments loaded straight from global per the documented VGPR layouts:
//   A lane(l): row M=l&15(+16), K runs [grp*8, +8) and [16+grp*8, +8)
//   B lane(l): row N=l&15,      K runs [grp*8, +8) and [16+grp*8, +8)
// ---------------------------------------------------------------------------
template <int MODE, bool HAS_BIAS>
__global__ __launch_bounds__(128)
void gemm_wmma_f16(const float* __restrict__ A, int lda,
                   const float* __restrict__ W,            // N x K row-major
                   const float* __restrict__ bias,
                   float* __restrict__ C, int ldc,
                   int N, int K)
{
    const int lane = threadIdx.x & 31;
    const int wave = threadIdx.x >> 5;
    const int am   = lane & 15;
    const int grp  = lane >> 4;
    const int m0   = blockIdx.y * 32;
    const int col  = blockIdx.x * 64 + (wave << 4) + am;

    const float* arow0 = A + (size_t)(m0 + am) * lda + (grp << 3);
    const float* arow1 = arow0 + (size_t)16 * lda;
    const float* wrow  = W + (size_t)col * K + (grp << 3);

    v8f acc0 = {}, acc1 = {};

    for (int k0 = 0; k0 < K; k0 += 32) {
        v4f a00 = *(const v4f*)(arow0 + k0);
        v4f a01 = *(const v4f*)(arow0 + k0 + 4);
        v4f a02 = *(const v4f*)(arow0 + k0 + 16);
        v4f a03 = *(const v4f*)(arow0 + k0 + 20);
        v4f a10 = *(const v4f*)(arow1 + k0);
        v4f a11 = *(const v4f*)(arow1 + k0 + 4);
        v4f a12 = *(const v4f*)(arow1 + k0 + 16);
        v4f a13 = *(const v4f*)(arow1 + k0 + 20);
        v4f w0  = *(const v4f*)(wrow  + k0);
        v4f w1  = *(const v4f*)(wrow  + k0 + 4);
        v4f w2  = *(const v4f*)(wrow  + k0 + 16);
        v4f w3  = *(const v4f*)(wrow  + k0 + 20);

        v16h af0, af1, bf;
        #pragma unroll
        for (int i = 0; i < 4; ++i) {
            af0[i]      = (_Float16)a00[i];
            af0[i + 4]  = (_Float16)a01[i];
            af0[i + 8]  = (_Float16)a02[i];
            af0[i + 12] = (_Float16)a03[i];
            af1[i]      = (_Float16)a10[i];
            af1[i + 4]  = (_Float16)a11[i];
            af1[i + 8]  = (_Float16)a12[i];
            af1[i + 12] = (_Float16)a13[i];
            bf[i]       = (_Float16)w0[i];
            bf[i + 4]   = (_Float16)w1[i];
            bf[i + 8]   = (_Float16)w2[i];
            bf[i + 12]  = (_Float16)w3[i];
        }
        acc0 = __builtin_amdgcn_wmma_f32_16x16x32_f16(
                   false, af0, false, bf, (short)0, acc0, false, false);
        acc1 = __builtin_amdgcn_wmma_f32_16x16x32_f16(
                   false, af1, false, bf, (short)0, acc1, false, false);
    }

    // Epilogue (C/D layout: VGPR v -> M = v + 8*grp, N = lane&15)
    const float bv = HAS_BIAS ? bias[col] : 0.0f;
    #pragma unroll
    for (int half = 0; half < 2; ++half) {
        v8f* accp = half ? &acc1 : &acc0;
        #pragma unroll
        for (int v = 0; v < 8; ++v) {
            int   rowg = m0 + (half << 4) + (grp << 3) + v;
            float val  = (*accp)[v] + bv;
            if (MODE == 1) {
                float sp = __logf(1.0f + __expf(val));
                val = (val > 20.0f) ? val : sp;
            }
            if (MODE == 2) {
                int bb = rowg >> 12;               // rowg / LSEQ
                int ll = rowg & (LSEQ - 1);
                C[((size_t)bb * N + col) * (size_t)LSEQ + ll] = val;
            } else {
                C[(size_t)rowg * ldc + col] = val;
            }
        }
    }
}

// ---------------------------------------------------------------------------
// Depthwise causal conv (D_CONV=4) + SiLU:  u = silu(conv(xz[:, :512]) + b)
// ---------------------------------------------------------------------------
__global__ __launch_bounds__(256)
void conv_silu(const float* __restrict__ xz, const float* __restrict__ cw,
               const float* __restrict__ cb, float* __restrict__ u, int total)
{
    int idx = blockIdx.x * blockDim.x + threadIdx.x;
    if (idx >= total) return;
    int d  = idx & (DINNER - 1);
    int bl = idx >> 9;                 // b*L + l
    int l  = bl & (LSEQ - 1);
    float acc = cb[d];
    #pragma unroll
    for (int k = 0; k < 4; ++k) {
        int ls = l - 3 + k;
        if (ls >= 0)
            acc += xz[(size_t)(bl - 3 + k) * 1024 + d] * cw[d * 4 + k];
    }
    u[idx] = acc / (1.0f + __expf(-acc));
}

// ---------------------------------------------------------------------------
// Wc = from_seq_w (256x256) @ out_proj_w (256x512)   (tiny, one-shot)
// ---------------------------------------------------------------------------
__global__ __launch_bounds__(256)
void fuse_out_proj(const float* __restrict__ F, const float* __restrict__ P,
                   float* __restrict__ Wc)
{
    int j = blockIdx.x * blockDim.x + threadIdx.x;   // 0..511
    int i = blockIdx.y;                              // 0..255
    float acc = 0.0f;
    for (int k = 0; k < DIMC; ++k)
        acc += F[i * DIMC + k] * P[(size_t)k * DINNER + j];
    Wc[(size_t)i * DINNER + j] = acc;
}

// ---------------------------------------------------------------------------
// Chunked selective scan, phase 1: per-chunk (prod dA, partial state)
// grid = (NCHUNK, B), block = 512 (one thread per d)
// ---------------------------------------------------------------------------
__global__ __launch_bounds__(512)
void scan_phase1(const float* __restrict__ xdbc, const float* __restrict__ delta,
                 const float* __restrict__ u, const float* __restrict__ A_log,
                 float* __restrict__ Ap, float* __restrict__ Bp)
{
    int b = blockIdx.y, c = blockIdx.x, d = threadIdx.x;
    float Ac[DSTATE], a[DSTATE], p[DSTATE];
    #pragma unroll
    for (int n = 0; n < DSTATE; ++n) {
        Ac[n] = -__expf(A_log[d * DSTATE + n]);
        a[n]  = 1.0f;
        p[n]  = 0.0f;
    }
    int l0 = c * CLEN;
    for (int t = 0; t < CLEN; ++t) {
        size_t row = (size_t)b * LSEQ + l0 + t;
        float dt = delta[row * DINNER + d];
        float ut = u[row * DINNER + d];
        float du = dt * ut;
        const float* Bt = xdbc + row * XSTR + 16;
        v4f bch[4];
        #pragma unroll
        for (int q = 0; q < 4; ++q) bch[q] = *(const v4f*)(Bt + 4 * q);
        #pragma unroll
        for (int n = 0; n < DSTATE; ++n) {
            float dA = __expf(dt * Ac[n]);
            a[n] *= dA;
            p[n]  = dA * p[n] + du * bch[n >> 2][n & 3];
        }
    }
    size_t base = (((size_t)b * NCHUNK + c) * DINNER + d) * DSTATE;
    #pragma unroll
    for (int n = 0; n < DSTATE; ++n) { Ap[base + n] = a[n]; Bp[base + n] = p[n]; }
}

// ---------------------------------------------------------------------------
// Phase 2: prefix over chunks; Ap[...] is overwritten with h_init per chunk
// ---------------------------------------------------------------------------
__global__ __launch_bounds__(256)
void scan_phase2(float* __restrict__ Ap, const float* __restrict__ Bp)
{
    int idx = blockIdx.x * blockDim.x + threadIdx.x;   // B*DINNER*DSTATE
    int n = idx & (DSTATE - 1);
    int d = (idx >> 4) & (DINNER - 1);
    int b = idx >> 13;
    float h = 0.0f;
    for (int c = 0; c < NCHUNK; ++c) {
        size_t off = (((size_t)b * NCHUNK + c) * DINNER + d) * DSTATE + n;
        float a  = Ap[off];
        float bb = Bp[off];
        Ap[off] = h;                 // h_init for chunk c
        h = a * h + bb;
    }
}

// ---------------------------------------------------------------------------
// Phase 3: replay chunks from h_init, emit y = (scan + u*D) * silu(z)
// ---------------------------------------------------------------------------
__global__ __launch_bounds__(512)
void scan_phase3(const float* __restrict__ xz, const float* __restrict__ u,
                 const float* __restrict__ xdbc, const float* __restrict__ delta,
                 const float* __restrict__ A_log, const float* __restrict__ Dv,
                 const float* __restrict__ Hinit, float* __restrict__ y)
{
    int b = blockIdx.y, c = blockIdx.x, d = threadIdx.x;
    float Ac[DSTATE], h[DSTATE];
    size_t base = (((size_t)b * NCHUNK + c) * DINNER + d) * DSTATE;
    #pragma unroll
    for (int n = 0; n < DSTATE; ++n) {
        Ac[n] = -__expf(A_log[d * DSTATE + n]);
        h[n]  = Hinit[base + n];
    }
    float Dd = Dv[d];
    int l0 = c * CLEN;
    for (int t = 0; t < CLEN; ++t) {
        size_t row = (size_t)b * LSEQ + l0 + t;
        float dt = delta[row * DINNER + d];
        float ut = u[row * DINNER + d];
        float du = dt * ut;
        const float* Bt = xdbc + row * XSTR + 16;
        v4f bch[4], cch[4];
        #pragma unroll
        for (int q = 0; q < 4; ++q) {
            bch[q] = *(const v4f*)(Bt + 4 * q);
            cch[q] = *(const v4f*)(Bt + 16 + 4 * q);
        }
        float yt = 0.0f;
        #pragma unroll
        for (int n = 0; n < DSTATE; ++n) {
            float dA = __expf(dt * Ac[n]);
            h[n] = dA * h[n] + du * bch[n >> 2][n & 3];
            yt   = fmaf(h[n], cch[n >> 2][n & 3], yt);
        }
        float zt = xz[row * 1024 + DINNER + d];
        float sz = zt / (1.0f + __expf(-zt));
        y[row * DINNER + d] = (yt + ut * Dd) * sz;
    }
}

// ---------------------------------------------------------------------------
extern "C" void kernel_launch(void* const* d_in, const int* in_sizes, int n_in,
                              void* d_out, int out_size, void* d_ws, size_t ws_size,
                              hipStream_t stream)
{
    (void)in_sizes; (void)n_in; (void)out_size; (void)ws_size;

    const float* x_feat     = (const float*)d_in[0];
    const float* to_seq_w   = (const float*)d_in[1];
    const float* to_seq_b   = (const float*)d_in[2];
    const float* in_proj_w  = (const float*)d_in[3];
    const float* conv_w     = (const float*)d_in[4];
    const float* conv_b     = (const float*)d_in[5];
    const float* x_proj_w   = (const float*)d_in[6];
    const float* dt_proj_w  = (const float*)d_in[7];
    const float* dt_proj_b  = (const float*)d_in[8];
    const float* A_log      = (const float*)d_in[9];
    const float* Dvec       = (const float*)d_in[10];
    const float* out_proj_w = (const float*)d_in[11];
    const float* from_seq_w = (const float*)d_in[12];
    const float* from_seq_b = (const float*)d_in[13];
    float* out = (float*)d_out;

    const size_t M = (size_t)NBATCH * LSEQ;            // 32768
    float* ws = (float*)d_ws;
    size_t off = 0;
    float* xz    = ws + off; off += M * 1024;          // u0 | z
    float* ubuf  = ws + off; off += M * DINNER;        // conv+silu output
    float* seq0  = ws + off;                           // transposed input
    float* seq   = seq0 + M * DIMC;                    // after to_seq
    float* deltb = seq0;                               // reuse (M*512) after GEMMs 1-2
    off += M * DINNER;
    float* xdbc  = ws + off; off += M * XSTR;
    float* ybuf  = ws + off; off += M * DINNER;
    float* Wc    = ws + off; off += (size_t)DIMC * DINNER;
    float* Wxp   = ws + off; off += (size_t)64 * DINNER;    // padded x_proj_w
    float* Wdt   = ws + off; off += (size_t)DINNER * 32;    // padded dt_proj_w
    float* Ap    = ws + off; off += (size_t)NBATCH * NCHUNK * DINNER * DSTATE;
    float* Bp    = ws + off; off += (size_t)NBATCH * NCHUNK * DINNER * DSTATE;

    // 0. Pad irregular weights; fuse output projections (independent, tiny)
    pad_xproj_w<<<(64 * DINNER) / 256, 256, 0, stream>>>(x_proj_w, Wxp);
    pad_dt_w<<<(DINNER * 32) / 256, 256, 0, stream>>>(dt_proj_w, Wdt);
    fuse_out_proj<<<dim3(DINNER / 256, DIMC), 256, 0, stream>>>(from_seq_w, out_proj_w, Wc);

    // 1. (B,C,L) -> (B*L, C)
    transpose_bcl_to_blc<<<dim3(LSEQ / 32, DIMC / 32, NBATCH), dim3(32, 8), 0, stream>>>(
        x_feat, seq0);

    // 2. seq = seq0 @ to_seq_w.T + to_seq_b
    gemm_wmma_f16<0, true><<<dim3(DIMC / 64, M / 32), 128, 0, stream>>>(
        seq0, DIMC, to_seq_w, to_seq_b, seq, DIMC, DIMC, DIMC);

    // 3. xz = seq @ in_proj_w.T
    gemm_wmma_f16<0, false><<<dim3(1024 / 64, M / 32), 128, 0, stream>>>(
        seq, DIMC, in_proj_w, nullptr, xz, 1024, 1024, DIMC);

    // 4. u = silu(depthwise_conv(xz[:, :512]) + conv_b)
    {
        int total = (int)(M * DINNER);
        conv_silu<<<(total + 255) / 256, 256, 0, stream>>>(xz, conv_w, conv_b, ubuf, total);
    }

    // 5. xdbc = u @ Wxp.T   (N padded to 64, stride 64; cols 48..63 zero)
    gemm_wmma_f16<0, false><<<dim3(1, M / 32), 128, 0, stream>>>(
        ubuf, DINNER, Wxp, nullptr, xdbc, XSTR, 64, DINNER);

    // 6. delta = softplus(xdbc[:, :32] @ Wdt.T + dt_proj_b)  (K padded to 32)
    gemm_wmma_f16<1, true><<<dim3(DINNER / 64, M / 32), 128, 0, stream>>>(
        xdbc, XSTR, Wdt, dt_proj_b, deltb, DINNER, DINNER, 32);

    // 7. Chunked selective scan (3 phases)
    scan_phase1<<<dim3(NCHUNK, NBATCH), DINNER, 0, stream>>>(
        xdbc, deltb, ubuf, A_log, Ap, Bp);
    scan_phase2<<<(NBATCH * DINNER * DSTATE) / 256, 256, 0, stream>>>(Ap, Bp);
    scan_phase3<<<dim3(NCHUNK, NBATCH), DINNER, 0, stream>>>(
        xz, ubuf, xdbc, deltb, A_log, Dvec, Ap, ybuf);

    // 8. out[b,c,l] = (y @ Wc.T + from_seq_b)[b*L+l, c]   (transposed store)
    gemm_wmma_f16<2, true><<<dim3(DIMC / 64, M / 32), 128, 0, stream>>>(
        ybuf, DINNER, Wc, from_seq_b, out, LSEQ, DIMC, DINNER);
}
